// EncoderLayerInnerAttention_42700564857556
// MI455X (gfx1250) — compile-verified, matrix operand order checked by reading, not run on hardware
//
#include <hip/hip_runtime.h>
#include <hip/hip_bf16.h>
#include <math.h>

typedef __attribute__((ext_vector_type(16))) _Float16 v16h;
typedef __attribute__((ext_vector_type(8)))  _Float16 v8h;
typedef __attribute__((ext_vector_type(8)))  float    v8f;

#define SEQ    1024
#define CMODEL 256
#define DFF_N  1024
#define NHEAD  8
#define HDIM   32
#define BTOT   8
#define MTOT   (BTOT * SEQ)   // 8192
#define LN_EPS 1e-5f
#define POS_W  0.1f

// ---------------- WMMA helpers ----------------

__device__ __forceinline__ v8f wmma16(v16h a, v16h b, v8f c) {
  // D = A(16x32 f16) * B(32x16 f16) + C(16x16 f32)
  return __builtin_amdgcn_wmma_f32_16x16x32_f16(false, a, false, b, (short)0, c,
                                                false, false);
}

__device__ __forceinline__ v8f zero8() {
  v8f z;
#pragma unroll
  for (int i = 0; i < 8; ++i) z[i] = 0.0f;
  return z;
}

// Load a 16x32 (MxK) A-fragment (or equivalently a B-fragment from the
// transposed-B matrix stored N-major) from row-major f16, row stride `ld`.
// Per ISA 16-bit A layout: lane L -> row (L&15); K = 8*(L>>4)+0..7 and
// 16+8*(L>>4)+0..7  => two contiguous 16-byte loads.
__device__ __forceinline__ v16h load_frag(const _Float16* __restrict__ base,
                                          int ld, int r0, int k0) {
  const int lane = threadIdx.x & 31;
  const int m  = r0 + (lane & 15);
  const int kb = k0 + ((lane >> 4) << 3);
  const _Float16* p = base + (size_t)m * (size_t)ld + kb;
  v8h lo = *(const v8h*)(p);
  v8h hi = *(const v8h*)(p + 16);
  v16h f;
#pragma unroll
  for (int i = 0; i < 8; ++i) { f[i] = lo[i]; f[i + 8] = hi[i]; }
  return f;
}

// ---------------- positional encoding + feature prep ----------------

__global__ void prep_feat(const float* __restrict__ x, const float* __restrict__ y,
                          float* __restrict__ feat32, _Float16* __restrict__ feat16) {
  int idx = blockIdx.x * blockDim.x + threadIdx.x;   // [0, MTOT*CMODEL)
  int c = idx & 255;
  int m = idx >> 8;
  int b = m >> 10;
  int s = m & 1023;
  int hh = s >> 5, ww = s & 31;
  const float* src = (b < 4) ? x : y;
  int bb = b & 3;
  float v = src[(size_t)(bb * 256 + c) * 1024 + s];

  int j = (c < 128) ? c : (c - 128);
  const float scale = 6.28318530717958647692f;
  float e = (c < 128) ? ((float)(hh + 1) / (32.0f + 1e-6f) * scale)
                      : ((float)(ww + 1) / (32.0f + 1e-6f) * scale);
  float expo = (float)(2 * (j >> 1)) * (1.0f / 128.0f);
  float t = __powf(10000.0f, expo);
  float p = e / t;
  float pe = (j & 1) ? __cosf(p) : __sinf(p);
  float f = v + POS_W * pe;
  feat32[idx] = f;
  feat16[idx] = (_Float16)f;
}

// ---------------- weight conversion to f16 ----------------

__global__ void cvt_w(const float* __restrict__ wqkv, const float* __restrict__ wout,
                      const float* __restrict__ w1, const float* __restrict__ w2,
                      _Float16* __restrict__ dst) {
  int i = blockIdx.x * blockDim.x + threadIdx.x;     // [0, 786432)
  float v;
  if (i < 196608)       v = wqkv[i];
  else if (i < 262144)  v = wout[i - 196608];
  else if (i < 524288)  v = w1[i - 262144];
  else                  v = w2[i - 524288];
  dst[i] = (_Float16)v;
}

// ---------------- QKV projection: [8192,256] x [256,768] ----------------

__global__ __launch_bounds__(32)
void gemm_qkv(const _Float16* __restrict__ A, const _Float16* __restrict__ W,
              const float* __restrict__ bias,
              _Float16* __restrict__ Qb, _Float16* __restrict__ Kb,
              _Float16* __restrict__ Vb) {
  const int row0 = blockIdx.x * 16;
  const int n0   = blockIdx.y * 64;
  const int lane = threadIdx.x & 31;
  v8f acc[4];
#pragma unroll
  for (int j = 0; j < 4; ++j) acc[j] = zero8();

  for (int k0 = 0; k0 < CMODEL; k0 += 32) {
    v16h a = load_frag(A, CMODEL, row0, k0);
#pragma unroll
    for (int j = 0; j < 4; ++j) {
      v16h b = load_frag(W, CMODEL, n0 + j * 16, k0);
      acc[j] = wmma16(a, b, acc[j]);
    }
  }

  const int mb = row0 + ((lane >> 4) << 3);
#pragma unroll
  for (int j = 0; j < 4; ++j) {
    int o = n0 + j * 16 + (lane & 15);
    float bv = bias[o];
    int part = o >> 8;          // 0=Q 1=K 2=V
    int h = (o & 255) >> 5;
    int d = o & 31;
#pragma unroll
    for (int r = 0; r < 8; ++r) {
      int m = mb + r;
      int b = m >> 10, s = m & 1023;
      _Float16 hv = (_Float16)(acc[j][r] + bv);
      size_t bh = (size_t)(b * NHEAD + h);
      if (part == 0)      Qb[(bh * SEQ + s) * HDIM + d] = hv;
      else if (part == 1) Kb[(bh * SEQ + s) * HDIM + d] = hv;
      else                Vb[(bh * HDIM + d) * SEQ + s] = hv;   // V transposed
    }
  }
}

// ---------------- attention: one wave per (b,h,q-tile of 16) ----------------

__global__ __launch_bounds__(32)
void attn_kernel(const _Float16* __restrict__ Qb, const _Float16* __restrict__ Kb,
                 const _Float16* __restrict__ Vb, _Float16* __restrict__ Ob) {
  __shared__ float sc[16 * 1024];   // 64 KB: full score row-block, in-place softmax
  const int gid = blockIdx.x;
  const int qt = gid & 63;
  const int bh = gid >> 6;
  const int b = bh >> 3, h = bh & 7;
  const int lane = threadIdx.x & 31;
  const int q0 = qt * 16;

  const _Float16* Q = Qb + (size_t)bh * SEQ * HDIM;
  const _Float16* K = Kb + (size_t)bh * SEQ * HDIM;
  const _Float16* V = Vb + (size_t)bh * HDIM * SEQ;

  // S = Q * K^T  (K dim = head dim = 32: one WMMA per 16-wide key tile)
  v16h qa = load_frag(Q, HDIM, q0, 0);
  for (int kt = 0; kt < 64; ++kt) {
    v16h kf = load_frag(K, HDIM, kt * 16, 0);
    v8f acc = zero8();
    acc = wmma16(qa, kf, acc);
    int n = kt * 16 + (lane & 15);
    int mb = (lane >> 4) << 3;
#pragma unroll
    for (int r = 0; r < 8; ++r) sc[(mb + r) * 1024 + n] = acc[r];
  }
  __syncthreads();

  // softmax over 1024 (scale 1/sqrt(32) folded into exp): lane -> row L&15, half L>>4
  {
    const int row = lane & 15;
    const int off = (lane >> 4) * 512;
    float* rp = sc + row * 1024 + off;
    float mx = -1e30f;
    for (int i = 0; i < 512; ++i) mx = fmaxf(mx, rp[i]);
    mx = fmaxf(mx, __shfl_xor(mx, 16, 32));
    const float isq = 0.17677669529663688110f;  // 1/sqrt(32)
    float sum = 0.0f;
    for (int i = 0; i < 512; ++i) {
      float p = __expf((rp[i] - mx) * isq);
      rp[i] = p;
      sum += p;
    }
    sum += __shfl_xor(sum, 16, 32);
    float inv = 1.0f / sum;
    for (int i = 0; i < 512; ++i) rp[i] *= inv;
  }
  __syncthreads();

  // O = P * V ; A-fragment read straight from LDS in fragment layout
  v8f o0 = zero8(), o1 = zero8();
  for (int kc = 0; kc < SEQ; kc += 32) {
    const int m = lane & 15;
    const int kb = kc + ((lane >> 4) << 3);
    const float* rp = sc + m * 1024 + kb;
    v16h a;
#pragma unroll
    for (int i = 0; i < 8; ++i) { a[i] = (_Float16)rp[i]; a[i + 8] = (_Float16)rp[i + 16]; }
    v16h b0 = load_frag(V, SEQ, 0, kc);    // Vt rows = d 0..15
    v16h b1 = load_frag(V, SEQ, 16, kc);   // Vt rows = d 16..31
    o0 = wmma16(a, b0, o0);
    o1 = wmma16(a, b1, o1);
  }

  const int mb = q0 + ((lane >> 4) << 3);
  const int dl = lane & 15;
#pragma unroll
  for (int r = 0; r < 8; ++r) {
    size_t m = (size_t)b * SEQ + mb + r;
    Ob[m * CMODEL + h * HDIM + dl]      = (_Float16)o0[r];
    Ob[m * CMODEL + h * HDIM + 16 + dl] = (_Float16)o1[r];
  }
}

// ---------------- out-projection + residual: [8192,256]x[256,256] ----------------

__global__ __launch_bounds__(32)
void gemm_out(const _Float16* __restrict__ A, const _Float16* __restrict__ W,
              const float* __restrict__ bias, const float* __restrict__ resid,
              float* __restrict__ R1) {
  const int row0 = blockIdx.x * 16;
  const int n0   = blockIdx.y * 64;
  const int lane = threadIdx.x & 31;
  v8f acc[4];
#pragma unroll
  for (int j = 0; j < 4; ++j) acc[j] = zero8();
  for (int k0 = 0; k0 < CMODEL; k0 += 32) {
    v16h a = load_frag(A, CMODEL, row0, k0);
#pragma unroll
    for (int j = 0; j < 4; ++j) {
      v16h b = load_frag(W, CMODEL, n0 + j * 16, k0);
      acc[j] = wmma16(a, b, acc[j]);
    }
  }
  const int mb = row0 + ((lane >> 4) << 3);
#pragma unroll
  for (int j = 0; j < 4; ++j) {
    int o = n0 + j * 16 + (lane & 15);
    float bv = bias[o];
#pragma unroll
    for (int r = 0; r < 8; ++r) {
      size_t idx = (size_t)(mb + r) * CMODEL + o;
      R1[idx] = resid[idx] + acc[j][r] + bv;
    }
  }
}

// ---------------- LayerNorm (in-place f32, optional f16 copy) ----------------

__global__ void ln_kernel(float* __restrict__ io, const float* __restrict__ g,
                          const float* __restrict__ bta, _Float16* __restrict__ out16) {
  const int wave = threadIdx.x >> 5;
  const int lane = threadIdx.x & 31;
  const size_t row = (size_t)blockIdx.x * 8 + wave;
  float* rp = io + row * CMODEL;
  float vals[8];
  float s = 0.0f, s2 = 0.0f;
#pragma unroll
  for (int i = 0; i < 8; ++i) {
    float v = rp[lane + 32 * i];
    vals[i] = v; s += v; s2 += v * v;
  }
#pragma unroll
  for (int msk = 16; msk >= 1; msk >>= 1) {
    s  += __shfl_xor(s,  msk, 32);
    s2 += __shfl_xor(s2, msk, 32);
  }
  float mu = s * (1.0f / 256.0f);
  float var = s2 * (1.0f / 256.0f) - mu * mu;
  float rinv = rsqrtf(var + LN_EPS);
#pragma unroll
  for (int i = 0; i < 8; ++i) {
    int c = lane + 32 * i;
    float yv = (vals[i] - mu) * rinv * g[c] + bta[c];
    rp[c] = yv;
    out16[row * CMODEL + c] = (_Float16)yv;
  }
}

// ---------------- FFN1: [8192,256]x[256,1024] + ReLU ----------------

__global__ __launch_bounds__(32)
void gemm_ffn1(const _Float16* __restrict__ A, const _Float16* __restrict__ W,
               const float* __restrict__ bias, _Float16* __restrict__ H) {
  const int row0 = blockIdx.x * 16;
  const int n0   = blockIdx.y * 64;
  const int lane = threadIdx.x & 31;
  v8f acc[4];
#pragma unroll
  for (int j = 0; j < 4; ++j) acc[j] = zero8();
  for (int k0 = 0; k0 < CMODEL; k0 += 32) {
    v16h a = load_frag(A, CMODEL, row0, k0);
#pragma unroll
    for (int j = 0; j < 4; ++j) {
      v16h b = load_frag(W, CMODEL, n0 + j * 16, k0);
      acc[j] = wmma16(a, b, acc[j]);
    }
  }
  const int mb = row0 + ((lane >> 4) << 3);
#pragma unroll
  for (int j = 0; j < 4; ++j) {
    int o = n0 + j * 16 + (lane & 15);
    float bv = bias[o];
#pragma unroll
    for (int r = 0; r < 8; ++r) {
      float v = fmaxf(acc[j][r] + bv, 0.0f);
      H[(size_t)(mb + r) * DFF_N + o] = (_Float16)v;
    }
  }
}

// ---------------- FFN2 + residual: [8192,1024]x[1024,256] ----------------

__global__ __launch_bounds__(32)
void gemm_ffn2(const _Float16* __restrict__ H, const _Float16* __restrict__ W,
               const float* __restrict__ bias, const float* __restrict__ resid,
               float* __restrict__ R2) {
  const int row0 = blockIdx.x * 16;
  const int n0   = blockIdx.y * 64;
  const int lane = threadIdx.x & 31;
  v8f acc[4];
#pragma unroll
  for (int j = 0; j < 4; ++j) acc[j] = zero8();
  for (int k0 = 0; k0 < DFF_N; k0 += 32) {
    v16h a = load_frag(H, DFF_N, row0, k0);
#pragma unroll
    for (int j = 0; j < 4; ++j) {
      v16h b = load_frag(W, DFF_N, n0 + j * 16, k0);
      acc[j] = wmma16(a, b, acc[j]);
    }
  }
  const int mb = row0 + ((lane >> 4) << 3);
#pragma unroll
  for (int j = 0; j < 4; ++j) {
    int o = n0 + j * 16 + (lane & 15);
    float bv = bias[o];
#pragma unroll
    for (int r = 0; r < 8; ++r) {
      size_t idx = (size_t)(mb + r) * CMODEL + o;
      R2[idx] = resid[idx] + acc[j][r] + bv;
    }
  }
}

// ---------------- LN2 + transpose back to (B,C,H,W) ----------------

__global__ void ln2_final(const float* __restrict__ R2, const float* __restrict__ g,
                          const float* __restrict__ bta, float* __restrict__ out) {
  const int wave = threadIdx.x >> 5;
  const int lane = threadIdx.x & 31;
  const size_t row = (size_t)blockIdx.x * 8 + wave;
  const float* rp = R2 + row * CMODEL;
  float vals[8];
  float s = 0.0f, s2 = 0.0f;
#pragma unroll
  for (int i = 0; i < 8; ++i) {
    float v = rp[lane + 32 * i];
    vals[i] = v; s += v; s2 += v * v;
  }
#pragma unroll
  for (int msk = 16; msk >= 1; msk >>= 1) {
    s  += __shfl_xor(s,  msk, 32);
    s2 += __shfl_xor(s2, msk, 32);
  }
  float mu = s * (1.0f / 256.0f);
  float var = s2 * (1.0f / 256.0f) - mu * mu;
  float rinv = rsqrtf(var + LN_EPS);

  int b = (int)(row >> 10);
  int sp = (int)(row & 1023);
  size_t obase = (b < 4) ? 0 : (size_t)4 * 256 * 1024;
  int bb = b & 3;
#pragma unroll
  for (int i = 0; i < 8; ++i) {
    int c = lane + 32 * i;
    float yv = (vals[i] - mu) * rinv * g[c] + bta[c];
    out[obase + (size_t)(bb * 256 + c) * 1024 + sp] = yv;
  }
}

// ---------------- host-side orchestration ----------------

extern "C" void kernel_launch(void* const* d_in, const int* in_sizes, int n_in,
                              void* d_out, int out_size, void* d_ws, size_t ws_size,
                              hipStream_t stream) {
  (void)in_sizes; (void)n_in; (void)out_size; (void)ws_size;

  const float* x          = (const float*)d_in[0];
  const float* y          = (const float*)d_in[1];
  /* d_in[2] = nhead (int) — fixed at 8 */
  const float* in_proj_w  = (const float*)d_in[3];
  const float* in_proj_b  = (const float*)d_in[4];
  const float* out_proj_w = (const float*)d_in[5];
  const float* out_proj_b = (const float*)d_in[6];
  const float* w1         = (const float*)d_in[7];
  const float* b1         = (const float*)d_in[8];
  const float* w2         = (const float*)d_in[9];
  const float* b2         = (const float*)d_in[10];
  const float* ln1_g      = (const float*)d_in[11];
  const float* ln1_b      = (const float*)d_in[12];
  const float* ln2_g      = (const float*)d_in[13];
  const float* ln2_b      = (const float*)d_in[14];

  char* ws = (char*)d_ws;
  float*    feat32 = (float*)(ws + 0);               // 8 MB (reused as R2)
  float*    R1     = (float*)(ws + 8388608);         // 8 MB (in-place LN1 -> ln1_f32)
  _Float16* feat16 = (_Float16*)(ws + 16777216);     // 4 MB
  _Float16* ln116  = (_Float16*)(ws + 20971520);     // 4 MB
  _Float16* w16    = (_Float16*)(ws + 25165824);     // 1.5 MB
  _Float16* Qb     = (_Float16*)(ws + 26738688);     // 4 MB
  _Float16* Kb     = (_Float16*)(ws + 30932992);     // 4 MB
  _Float16* Vb     = (_Float16*)(ws + 35127296);     // 4 MB (transposed)
  _Float16* Ob     = (_Float16*)(ws + 39321600);     // 4 MB
  _Float16* Hb     = (_Float16*)(ws + 43515904);     // 16 MB

  _Float16* wqkv16 = w16;
  _Float16* wout16 = w16 + 196608;
  _Float16* w116   = w16 + 262144;
  _Float16* w216   = w16 + 524288;
  float*    R2     = feat32;  // feat32 dead after gemm_out

  prep_feat<<<8192, 256, 0, stream>>>(x, y, feat32, feat16);
  cvt_w<<<3072, 256, 0, stream>>>(in_proj_w, out_proj_w, w1, w2, w16);
  gemm_qkv<<<dim3(512, 12), 32, 0, stream>>>(feat16, wqkv16, in_proj_b, Qb, Kb, Vb);
  attn_kernel<<<4096, 32, 0, stream>>>(Qb, Kb, Vb, Ob);
  gemm_out<<<dim3(512, 4), 32, 0, stream>>>(Ob, wout16, out_proj_b, feat32, R1);
  ln_kernel<<<1024, 256, 0, stream>>>(R1, ln1_g, ln1_b, ln116);
  gemm_ffn1<<<dim3(512, 16), 32, 0, stream>>>(ln116, w116, b1, Hb);
  gemm_ffn2<<<dim3(512, 4), 32, 0, stream>>>(Hb, w216, b2, R1, R2);
  ln2_final<<<1024, 256, 0, stream>>>(R2, ln2_g, ln2_b, (float*)d_out);
}
